// TransformerBlockLocalEncode_84198538871233
// MI455X (gfx1250) — matrix operand order repaired and checked by reading, not compiled, not run on hardware
//
#include <hip/hip_runtime.h>

// ---------------------------------------------------------------------------
// CDNA5 (gfx1250) transformer block w/ windowed attention + ToMe merge + MLP.
// GEMMs via v_wmma_f32_16x16x32_f16; A-tiles staged by the Tensor Data Mover
// (tensor_load_to_lds + s_wait_tensorcnt), B-tiles by async global->LDS
// (global_load_async_to_lds_b128 + s_wait_asynccnt), with manual fallbacks.
// ---------------------------------------------------------------------------

typedef __attribute__((ext_vector_type(16))) _Float16 v16h;
typedef __attribute__((ext_vector_type(8)))  float    v8f;
typedef __attribute__((ext_vector_type(8)))  _Float16 h8v;
typedef __attribute__((ext_vector_type(4)))  unsigned int u32x4;
typedef __attribute__((ext_vector_type(8)))  int i32x8;
typedef __attribute__((ext_vector_type(4)))  int i32x4;

#define AS1 __attribute__((address_space(1)))
#define AS3 __attribute__((address_space(3)))

#ifndef __has_builtin
#define __has_builtin(x) 0
#endif
#if __has_builtin(__builtin_amdgcn_global_load_async_to_lds_b128) && \
    __has_builtin(__builtin_amdgcn_s_wait_asynccnt)
#define HAVE_ASYNC 1
#else
#define HAVE_ASYNC 0
#endif
#if __has_builtin(__builtin_amdgcn_tensor_load_to_lds) && \
    __has_builtin(__builtin_amdgcn_s_wait_tensorcnt)
#define HAVE_TDM 1
#else
#define HAVE_TDM 0
#endif

#define WMMA(a,b,c) __builtin_amdgcn_wmma_f32_16x16x32_f16(false,(a),false,(b),(short)0,(c),false,false)

#define B_  2
#define L_  2048
#define D_  768
#define H_  12
#define HD_ 64
#define TA_ 1024   // tokens in a/b sets
#define R_  512    // merged count
#define L2_ 1536   // tokens after merge
#define HID_ 3072

// ---- WMMA fragment loaders (wave32) ---------------------------------------
// A fragment: 16x32 f16, row m = lane&15, K range 0..15 (lanes 0-15) / 16..31 (lanes 16-31)
static __device__ __forceinline__ v16h load_a_frag(const _Float16* base, int lda) {
  const int l = threadIdx.x & 31;
  const int m = l & 15;
  const int kb = (l < 16) ? 0 : 16;
  const _Float16* p = base + m * lda + kb;
  v16h a;
#pragma unroll
  for (int j = 0; j < 8; ++j) { a[j] = p[j]; a[j + 8] = p[8 + j]; }
  return a;
}
// B fragment: 32x16 f16 from row-major [K][N] tile; col n = n0+(lane&15),
// K range 0..15 (lanes 0-15) / 16..31 (lanes 16-31)
static __device__ __forceinline__ v16h load_b_frag(const _Float16* base, int ldb, int n0) {
  const int l = threadIdx.x & 31;
  const int n = n0 + (l & 15);
  const int kb = (l < 16) ? 0 : 16;
  v16h b;
#pragma unroll
  for (int e = 0; e < 16; ++e) b[e] = base[(kb + e) * ldb + n];
  return b;
}
// B fragment of M^T where M is row-major [N][ldm]: b[k][n] = M[n][k] (contiguous per lane)
static __device__ __forceinline__ v16h load_bt_frag(const _Float16* base, int ldm) {
  const int l = threadIdx.x & 31;
  const int n = l & 15;
  const int kb = (l < 16) ? 0 : 16;
  const _Float16* p = base + n * ldm + kb;
  v16h b;
#pragma unroll
  for (int e = 0; e < 16; ++e) b[e] = p[e];
  return b;
}

#if HAVE_TDM
// LDS byte offset (workgroup-relative) of a generic pointer into __shared__
static __device__ __forceinline__ unsigned lds_off_u32(const void* p) {
  return (unsigned)(unsigned long long)(AS3 const char*)p;
}
// TDM: load a 2D tile of `rows` x 32 halfs from row-major src (row stride ldK
// halfs) into LDS at ldsOff, inserting 4 DWORDs of padding after every
// 16 DWORDs stored (i.e. LDS row stride = 40 halfs). One issue per wave.
static __device__ __forceinline__ void tdm_load_a_tile(const _Float16* src, int ldK,
                                                       int rows, int dimM,
                                                       unsigned ldsOff) {
  const unsigned long long ga = (unsigned long long)(const void*)src;
  u32x4 g0;
  g0[0] = 1u;                                   // count=1, user descriptor
  g0[1] = ldsOff;                               // lds_addr
  g0[2] = (unsigned)ga;                         // global_addr[31:0]
  g0[3] = (unsigned)((ga >> 32) & 0x01FFFFFFu) | 0x80000000u;  // addr[56:32] | type=2
  i32x8 g1;
  // data_size=2B (code 1), pad_enable, pad_interval=16 DW (code 3), pad_amount=4 DW (code 3)
  g1[0] = (int)(0x10000u | 0x100000u | (3u << 22) | (3u << 25));
  g1[1] = (int)(((unsigned)ldK & 0xFFFFu) << 16);                    // tensor_dim0[15:0]
  g1[2] = (int)((((unsigned)ldK >> 16) & 0xFFFFu) |
                (((unsigned)dimM & 0xFFFFu) << 16));                 // dim0[31:16] | dim1[15:0]
  g1[3] = (int)((((unsigned)dimM >> 16) & 0xFFFFu) | (32u << 16));   // dim1[31:16] | tile_dim0=32
  g1[4] = rows;                                                      // tile_dim1 | tile_dim2=0
  g1[5] = ldK;                                                       // tensor_dim0_stride[31:0]
  g1[6] = 0;                                                         // stride[47:32] | dim1_stride lo
  g1[7] = 0;
  const i32x4 gz = {0, 0, 0, 0};
#if __clang_major__ >= 23
  const i32x8 gz8 = {0, 0, 0, 0, 0, 0, 0, 0};
  __builtin_amdgcn_tensor_load_to_lds(g0, g1, gz, gz, gz8, 0);
#else
  __builtin_amdgcn_tensor_load_to_lds(g0, g1, gz, gz, 0);
#endif
}
#endif

// ---- Generic tiled WMMA GEMM: C[M,N] = A16[M,K] @ B16[K,N] + bias ----------
// EPI: 0 = f32 store (+bias), 1 = f32 +bias+residual, 2 = gelu(acc+bias)->f16
// Block: 256 threads = 8 waves; tile 64x128, K-step 32; wave (wm 0..3, wn 0..1),
// each wave computes 16x64 (1 A-frag x 4 B-frags x 4 WMMA per K-step).
template <int EPI>
__global__ __launch_bounds__(256) void gemm_wmma(
    const _Float16* __restrict__ A, const _Float16* __restrict__ B,
    const float* __restrict__ bias, const float* __restrict__ res,
    void* __restrict__ Cout, int M, int N, int K,
    long sAb, long sBb, long sCb) {
  const int z = blockIdx.z;
  A += (long)z * sAb;
  B += (long)z * sBb;
  const int n0 = blockIdx.x * 128, m0 = blockIdx.y * 64;
  const int t = threadIdx.x;
  const int wave = t >> 5;
  const int wm = wave & 3, wn = wave >> 2;
  __shared__ __align__(16) _Float16 As[64 * 40];   // 64 rows x 32 (+8 pad)
  __shared__ __align__(16) _Float16 Bs[32 * 136];  // 32 rows x 128 (+8 pad)
  const v8f vz = {0.f, 0.f, 0.f, 0.f, 0.f, 0.f, 0.f, 0.f};
  v8f acc[4];
  acc[0] = vz; acc[1] = vz; acc[2] = vz; acc[3] = vz;
  const int arow = t >> 2, acol = (t & 3) * 8;   // A staging: 1 x h8 per thread
  const int brow = t >> 3, bcol = (t & 7) * 16;  // B staging: 2 x h8 per thread
#if HAVE_TDM
  const unsigned asOff = lds_off_u32(&As[0]);
#endif
  for (int k0 = 0; k0 < K; k0 += 32) {
    // ---- stage A tile (64 x 32 halfs -> As[64][40]) ----
#if HAVE_TDM
    if (wave == 0) tdm_load_a_tile(&A[(long)m0 * K + k0], K, 64, M, asOff);
#elif HAVE_ASYNC
    __builtin_amdgcn_global_load_async_to_lds_b128(
        (AS1 i32x4*)&A[(long)(m0 + arow) * K + k0 + acol],
        (AS3 i32x4*)&As[arow * 40 + acol], 0, 0);
#else
    *(h8v*)&As[arow * 40 + acol] = *(const h8v*)&A[(long)(m0 + arow) * K + k0 + acol];
#endif
    // ---- stage B tile (32 x 128 halfs -> Bs[32][136]) ----
#if HAVE_ASYNC
    __builtin_amdgcn_global_load_async_to_lds_b128(
        (AS1 i32x4*)&B[(long)(k0 + brow) * N + n0 + bcol],
        (AS3 i32x4*)&Bs[brow * 136 + bcol], 0, 0);
    __builtin_amdgcn_global_load_async_to_lds_b128(
        (AS1 i32x4*)&B[(long)(k0 + brow) * N + n0 + bcol + 8],
        (AS3 i32x4*)&Bs[brow * 136 + bcol + 8], 0, 0);
#else
    *(h8v*)&Bs[brow * 136 + bcol] =
        *(const h8v*)&B[(long)(k0 + brow) * N + n0 + bcol];
    *(h8v*)&Bs[brow * 136 + bcol + 8] =
        *(const h8v*)&B[(long)(k0 + brow) * N + n0 + bcol + 8];
#endif
    if (k0 + 32 < K)
      __builtin_prefetch(&B[(long)(k0 + 32 + brow) * N + n0 + bcol], 0, 1);
#if HAVE_ASYNC
    __builtin_amdgcn_s_wait_asynccnt(0);
#endif
#if HAVE_TDM
    if (wave == 0) __builtin_amdgcn_s_wait_tensorcnt(0);
#endif
    __syncthreads();
    const v16h a = load_a_frag(&As[(wm * 16) * 40], 40);
#pragma unroll
    for (int u = 0; u < 4; ++u)
      acc[u] = WMMA(a, load_b_frag(Bs, 136, wn * 64 + u * 16), acc[u]);
    __syncthreads();
  }
  const int l  = t & 31;
  const int nl = l & 15;
  const int rb = (l < 16) ? 0 : 8;
  const int c0 = n0 + wn * 64 + nl;
  float bv[4];
#pragma unroll
  for (int u = 0; u < 4; ++u) bv[u] = bias ? bias[c0 + u * 16] : 0.f;
#pragma unroll
  for (int i = 0; i < 8; ++i) {
    const long row = m0 + wm * 16 + rb + i;
#pragma unroll
    for (int u = 0; u < 4; ++u) {
      const long c = c0 + u * 16;
      float xv = acc[u][i] + bv[u];
      if (EPI == 0) {
        float* C = (float*)Cout + (long)z * sCb;
        C[row * N + c] = xv;
      } else if (EPI == 1) {
        float* C = (float*)Cout;
        C[row * N + c] = xv + res[row * (long)N + c];
      } else {
        _Float16* C = (_Float16*)Cout;
        xv = 0.5f * xv * (1.f + erff(xv * 0.70710678118654752f));
        C[row * N + c] = (_Float16)xv;
      }
    }
  }
}

// ---- LayerNorm -> f16 ------------------------------------------------------
__global__ __launch_bounds__(256) void ln_f16(const float* __restrict__ x,
                                              const float* __restrict__ g,
                                              const float* __restrict__ bb,
                                              _Float16* __restrict__ out, int D) {
  const int row = blockIdx.x;
  const int t = threadIdx.x;
  const float* xr = x + (long)row * D;
  __shared__ float red[8];
  float s = 0.f;
  for (int c = t; c < D; c += 256) s += xr[c];
#pragma unroll
  for (int m = 16; m; m >>= 1) s += __shfl_xor(s, m, 32);
  if ((t & 31) == 0) red[t >> 5] = s;
  __syncthreads();
  float tot = 0.f;
#pragma unroll
  for (int w = 0; w < 8; ++w) tot += red[w];
  const float mean = tot / D;
  __syncthreads();
  float v = 0.f;
  for (int c = t; c < D; c += 256) { const float d = xr[c] - mean; v += d * d; }
#pragma unroll
  for (int m = 16; m; m >>= 1) v += __shfl_xor(v, m, 32);
  if ((t & 31) == 0) red[t >> 5] = v;
  __syncthreads();
  float tv = 0.f;
#pragma unroll
  for (int w = 0; w < 8; ++w) tv += red[w];
  const float rstd = rsqrtf(tv / D + 1e-5f);
  _Float16* o = out + (long)row * D;
  for (int c = t; c < D; c += 256)
    o[c] = (_Float16)((xr[c] - mean) * rstd * g[c] + bb[c]);
}

// ---- f32 -> f16 conversion -------------------------------------------------
__global__ void cvt_f16(const float* __restrict__ src, _Float16* __restrict__ dst, long n) {
  const long i = (long)blockIdx.x * 256 + threadIdx.x;
  if (i < n) dst[i] = (_Float16)src[i];
}

// ---- split qkv f32 -> per-head f16 [B][H][L][64] --------------------------
__global__ void split_qkv(const float* __restrict__ qkv, _Float16* __restrict__ q16,
                          _Float16* __restrict__ k16, _Float16* __restrict__ v16) {
  const long i = (long)blockIdx.x * 256 + threadIdx.x;
  if (i >= (long)B_ * L_ * D_) return;
  const int c = (int)(i % D_);
  const long bl = i / D_;
  const int b = (int)(bl >> 11), l = (int)(bl & (L_ - 1));
  const int h = c >> 6, d = c & 63;
  const long dst = (((long)(b * H_ + h) * L_) + l) * HD_ + d;
  const float* src = qkv + bl * (3 * D_);
  q16[dst] = (_Float16)src[c];
  k16[dst] = (_Float16)src[D_ + c];
  v16[dst] = (_Float16)src[2 * D_ + c];
}

__global__ void prep_logts(const float* __restrict__ ts, float* __restrict__ lts, int n) {
  const int i = blockIdx.x * 256 + threadIdx.x;
  if (i < n) lts[i] = logf(fmaxf(ts[i], 1e-8f));
}

// ---- Windowed flash attention: 1 wave per (b,h,16-query tile) --------------
__global__ __launch_bounds__(32) void attn_kernel(
    const _Float16* __restrict__ Qg, const _Float16* __restrict__ Kg,
    const _Float16* __restrict__ Vg, const float* __restrict__ logts,
    const unsigned char* __restrict__ kpm, _Float16* __restrict__ outh) {
  const int q0 = blockIdx.x * 16;
  const int h = blockIdx.y, b = blockIdx.z;
  const int l = threadIdx.x;
  const long bh = ((long)(b * H_ + h)) * L_ * HD_;
  const _Float16* Q = Qg + bh;
  const _Float16* Kk = Kg + bh;
  const _Float16* V = Vg + bh;
  const v16h aq0 = load_a_frag(Q + (long)q0 * HD_, HD_);
  const v16h aq1 = load_a_frag(Q + (long)q0 * HD_ + 32, HD_);
  const v8f vz = {0.f, 0.f, 0.f, 0.f, 0.f, 0.f, 0.f, 0.f};
  v8f acc[4];
  acc[0] = vz; acc[1] = vz; acc[2] = vz; acc[3] = vz;
  float mrun = -INFINITY, lrun = 0.f;
  const int r = l & 15, cb = (l >> 4) * 16;
  const int rb = (l < 16) ? 0 : 8;
  const int nl = l & 15;
  __shared__ float St[16 * 32];
  __shared__ __align__(16) _Float16 Pt[16 * 32];
  __shared__ float raux[16];
  int lo = q0 - 128;  if (lo < 0) lo = 0;
  int hi = q0 + 16 + 128;  if (hi > L_) hi = L_;
  for (int kc = (lo & ~31); kc < hi; kc += 32) {
    v8f s0 = vz, s1 = vz;
    s0 = WMMA(aq0, load_bt_frag(Kk + (long)kc * HD_,       HD_), s0);
    s0 = WMMA(aq1, load_bt_frag(Kk + (long)kc * HD_ + 32,  HD_), s0);
    s1 = WMMA(aq0, load_bt_frag(Kk + (long)(kc + 16) * HD_,      HD_), s1);
    s1 = WMMA(aq1, load_bt_frag(Kk + (long)(kc + 16) * HD_ + 32, HD_), s1);
#pragma unroll
    for (int i = 0; i < 8; ++i) {
      St[(rb + i) * 32 + nl]      = s0[i];
      St[(rb + i) * 32 + 16 + nl] = s1[i];
    }
    __syncthreads();
    float p[16];
    float mloc = -INFINITY;
#pragma unroll
    for (int c = 0; c < 16; ++c) {
      const int key = kc + cb + c;
      const int qi = q0 + r;
      const bool bad = (key - qi > 128) || (qi - key > 128) || (kpm[b * L_ + key] != 0);
      const float v = bad ? -INFINITY : (St[r * 32 + cb + c] * 0.125f + logts[b * L_ + key]);
      p[c] = v;
      mloc = fmaxf(mloc, v);
    }
    mloc = fmaxf(mloc, __shfl_xor(mloc, 16, 32));
    const float mnew = fmaxf(mrun, mloc);
    const float scale = (mnew == -INFINITY) ? 1.f : __expf(mrun - mnew);
    float lloc = 0.f;
#pragma unroll
    for (int c = 0; c < 16; ++c) {
      const float e = (p[c] == -INFINITY) ? 0.f : __expf(p[c] - mnew);
      lloc += e;
      Pt[r * 32 + cb + c] = (_Float16)e;
    }
    lloc += __shfl_xor(lloc, 16, 32);
    lrun = lrun * scale + lloc;
    mrun = mnew;
    if (l < 16) raux[r] = scale;
    __syncthreads();
#pragma unroll
    for (int i = 0; i < 8; ++i) {
      const float f = raux[rb + i];
      acc[0][i] = acc[0][i] * f;
      acc[1][i] = acc[1][i] * f;
      acc[2][i] = acc[2][i] * f;
      acc[3][i] = acc[3][i] * f;
    }
    const v16h ap = load_a_frag(Pt, 32);
#pragma unroll
    for (int tt = 0; tt < 4; ++tt)
      acc[tt] = WMMA(ap, load_b_frag(V + (long)kc * HD_, HD_, tt * 16), acc[tt]);
    __syncthreads();
  }
  if (l < 16) raux[r] = 1.f / lrun;
  __syncthreads();
#pragma unroll
  for (int tt = 0; tt < 4; ++tt)
#pragma unroll
    for (int i = 0; i < 8; ++i) {
      const long row = (long)b * L_ + q0 + rb + i;
      outh[row * D_ + h * HD_ + tt * 16 + nl] = (_Float16)(acc[tt][i] * raux[rb + i]);
    }
}

// ---- k head-mean + normalize -> a16 (even tokens) and bT16 [64][1024] ------
__global__ __launch_bounds__(64) void kmean_norm(const float* __restrict__ qkv,
                                                 _Float16* __restrict__ a16,
                                                 _Float16* __restrict__ bT16) {
  const int bl = blockIdx.x;           // 0..B*L-1
  const int b = bl / L_, l = bl % L_;
  const int d = threadIdx.x;           // 0..63
  const float* base = qkv + (long)bl * (3 * D_) + D_ + d;
  float s = 0.f;
#pragma unroll
  for (int h = 0; h < H_; ++h) s += base[h * HD_];
  s *= (1.f / H_);
  __shared__ float sq[64];
  sq[d] = s * s;
  __syncthreads();
  for (int off = 32; off > 0; off >>= 1) {
    if (d < off) sq[d] += sq[d + off];
    __syncthreads();
  }
  const float kn = s / (sqrtf(sq[0]) + 1e-8f);
  if ((l & 1) == 0)
    a16[((long)b * TA_ + (l >> 1)) * HD_ + d] = (_Float16)kn;
  else
    bT16[((long)b * HD_ + d) * TA_ + (l >> 1)] = (_Float16)kn;
}

__global__ void set_prot(float* __restrict__ scores) {
  const int i = blockIdx.x * 256 + threadIdx.x;  // over B*TA
  if (i >= B_ * TA_) return;
  const int b = i >> 10, c = i & (TA_ - 1);
  scores[((long)b * TA_ + 0) * TA_ + c] = -INFINITY;
}

// ---- per-row max + argmax over scores --------------------------------------
__global__ __launch_bounds__(256) void rowargmax(const float* __restrict__ scores,
                                                 float* __restrict__ nmax,
                                                 int* __restrict__ nidx) {
  const int row = blockIdx.x;  // B*TA
  const float* rp = scores + (long)row * TA_;
  const int t = threadIdx.x;
  float best = -INFINITY;
  int bi = 0;
  for (int c = t; c < TA_; c += 256) {
    const float v = rp[c];
    if (v > best || (v == best && c < bi)) { best = v; bi = c; }
  }
  __shared__ float bv[256];
  __shared__ int bx[256];
  bv[t] = best; bx[t] = bi;
  __syncthreads();
  for (int off = 128; off > 0; off >>= 1) {
    if (t < off) {
      const float v2 = bv[t + off];
      const int i2 = bx[t + off];
      if (v2 > bv[t] || (v2 == bv[t] && i2 < bx[t])) { bv[t] = v2; bx[t] = i2; }
    }
    __syncthreads();
  }
  if (t == 0) { nmax[row] = bv[0]; nidx[row] = bx[0]; }
}

// ---- bitonic argsort (descending by node_max, ties by index) + unm sort ----
__global__ __launch_bounds__(1024) void tome_sort(const float* __restrict__ nmax,
                                                  const int* __restrict__ nidx,
                                                  int* __restrict__ srcI,
                                                  int* __restrict__ dstI,
                                                  int* __restrict__ unmI) {
  const int b = blockIdx.x;
  const int t = threadIdx.x;
  __shared__ float sk[TA_];
  __shared__ int si[TA_];
  __shared__ int su[R_];
  sk[t] = nmax[b * TA_ + t];
  si[t] = t;
  __syncthreads();
  for (int k = 2; k <= TA_; k <<= 1)
    for (int j = k >> 1; j > 0; j >>= 1) {
      const int ixj = t ^ j;
      if (ixj > t) {
        const bool up = ((t & k) == 0);
        const float ka = sk[t], kc = sk[ixj];
        const int ia = si[t], ic = si[ixj];
        const bool beforeAC = (ka > kc) || (ka == kc && ia < ic);
        if (up ? !beforeAC : beforeAC) {
          sk[t] = kc; sk[ixj] = ka;
          si[t] = ic; si[ixj] = ia;
        }
      }
      __syncthreads();
    }
  if (t < R_) {
    const int s = si[t];
    srcI[b * R_ + t] = s;
    dstI[b * R_ + t] = nidx[b * TA_ + s];
    su[t] = si[R_ + t];
  }
  __syncthreads();
  for (int k = 2; k <= R_; k <<= 1)
    for (int j = k >> 1; j > 0; j >>= 1) {
      const int ixj = t ^ j;
      if (t < R_ && ixj > t && ixj < R_) {
        const bool up = ((t & k) == 0);
        const int a = su[t], c = su[ixj];
        if (up ? (a > c) : (a < c)) { su[t] = c; su[ixj] = a; }
      }
      __syncthreads();
    }
  if (t < R_) unmI[b * R_ + t] = su[t];
}

// ---- merge: init rows (unm gather + dst copy) ------------------------------
__global__ __launch_bounds__(256) void merge_init(const float* __restrict__ xm,
                                                  const float* __restrict__ ts,
                                                  const int* __restrict__ unmI,
                                                  float* __restrict__ xsum,
                                                  float* __restrict__ snew) {
  const int i = blockIdx.x;  // 0..L2-1
  const int b = blockIdx.y;
  const int t = threadIdx.x;
  int srcl;
  if (i < R_) srcl = 2 * unmI[b * R_ + i];
  else        srcl = 2 * (i - R_) + 1;
  const float tsv = ts[b * L_ + srcl];
  const float* src = xm + ((long)b * L_ + srcl) * D_;
  float* dst = xsum + ((long)b * L2_ + i) * D_;
  for (int c = t; c < D_; c += 256) dst[c] = src[c] * tsv;
  if (t == 0) snew[b * L2_ + i] = tsv;
}

// ---- merge: scatter-add src rows into dst rows -----------------------------
__global__ __launch_bounds__(256) void merge_scatter(const float* __restrict__ xm,
                                                     const float* __restrict__ ts,
                                                     const int* __restrict__ srcI,
                                                     const int* __restrict__ dstI,
                                                     float* __restrict__ xsum,
                                                     float* __restrict__ snew) {
  const int j = blockIdx.x;  // 0..R-1
  const int b = blockIdx.y;
  const int t = threadIdx.x;
  const int srcl = 2 * srcI[b * R_ + j];
  const int d = dstI[b * R_ + j];
  const float tsv = ts[b * L_ + srcl];
  const float* src = xm + ((long)b * L_ + srcl) * D_;
  float* dst = xsum + ((long)b * L2_ + R_ + d) * D_;
  for (int c = t; c < D_; c += 256) atomicAdd(&dst[c], src[c] * tsv);
  if (t == 0) atomicAdd(&snew[b * L2_ + R_ + d], tsv);
}

__global__ __launch_bounds__(256) void merge_final(float* __restrict__ xsum,
                                                   const float* __restrict__ snew) {
  const int row = blockIdx.x;  // B*L2
  const int t = threadIdx.x;
  const float inv = 1.f / (snew[row] + 1e-8f);
  float* p = xsum + (long)row * D_;
  for (int c = t; c < D_; c += 256) p[c] *= inv;
}

// ---------------------------------------------------------------------------
extern "C" void kernel_launch(void* const* d_in, const int* in_sizes, int n_in,
                              void* d_out, int out_size, void* d_ws, size_t ws_size,
                              hipStream_t stream) {
  (void)in_sizes; (void)n_in; (void)out_size; (void)ws_size;
  const float* x     = (const float*)d_in[0];
  const float* ts    = (const float*)d_in[1];
  const float* ln1g  = (const float*)d_in[2];
  const float* ln1b  = (const float*)d_in[3];
  const float* ln2g  = (const float*)d_in[4];
  const float* ln2b  = (const float*)d_in[5];
  const float* Wqkv  = (const float*)d_in[6];
  const float* bqkv  = (const float*)d_in[7];
  const float* Wproj = (const float*)d_in[8];
  const float* bproj = (const float*)d_in[9];
  const float* W1    = (const float*)d_in[10];
  const float* b1    = (const float*)d_in[11];
  const float* W2    = (const float*)d_in[12];
  const float* b2    = (const float*)d_in[13];
  const unsigned char* kpm = (const unsigned char*)d_in[14];

  char* base = (char*)d_ws;
  size_t off = 0;
  auto alloc = [&](size_t bytes) -> void* {
    void* p = base + off;
    off += (bytes + 255) & ~(size_t)255;
    return p;
  };
  const long BL = (long)B_ * L_;          // 4096
  _Float16* h16    = (_Float16*)alloc(BL * D_ * 2);
  _Float16* wqkv16 = (_Float16*)alloc((size_t)D_ * 3 * D_ * 2);
  _Float16* wproj16= (_Float16*)alloc((size_t)D_ * D_ * 2);
  _Float16* w1_16  = (_Float16*)alloc((size_t)D_ * HID_ * 2);
  _Float16* w2_16  = (_Float16*)alloc((size_t)HID_ * D_ * 2);
  float*    qkvF   = (float*)   alloc(BL * 3 * D_ * 4);
  _Float16* q16    = (_Float16*)alloc(BL * D_ * 2);
  _Float16* k16    = (_Float16*)alloc(BL * D_ * 2);
  _Float16* v16    = (_Float16*)alloc(BL * D_ * 2);
  float*    lts    = (float*)   alloc(BL * 4);
  _Float16* attn16 = (_Float16*)alloc(BL * D_ * 2);
  float*    xmF    = (float*)   alloc(BL * D_ * 4);
  _Float16* a16    = (_Float16*)alloc((size_t)B_ * TA_ * HD_ * 2);
  _Float16* bT16   = (_Float16*)alloc((size_t)B_ * HD_ * TA_ * 2);
  float*    scores = (float*)   alloc((size_t)B_ * TA_ * TA_ * 4);
  float*    nmax   = (float*)   alloc((size_t)B_ * TA_ * 4);
  int*      nidx   = (int*)     alloc((size_t)B_ * TA_ * 4);
  int*      srcI   = (int*)     alloc((size_t)B_ * R_ * 4);
  int*      dstI   = (int*)     alloc((size_t)B_ * R_ * 4);
  int*      unmI   = (int*)     alloc((size_t)B_ * R_ * 4);
  float*    xsum   = (float*)   alloc((size_t)B_ * L2_ * D_ * 4);
  float*    snew   = (float*)   alloc((size_t)B_ * L2_ * 4);
  _Float16* h2_16  = (_Float16*)alloc((size_t)B_ * L2_ * D_ * 2);
  _Float16* mlp16  = (_Float16*)alloc((size_t)B_ * L2_ * HID_ * 2);

  // 1. LN1 -> f16, weight conversions
  ln_f16<<<(int)BL, 256, 0, stream>>>(x, ln1g, ln1b, h16, D_);
  cvt_f16<<<(int)(((long)D_ * 3 * D_ + 255) / 256), 256, 0, stream>>>(Wqkv, wqkv16, (long)D_ * 3 * D_);
  cvt_f16<<<(int)(((long)D_ * D_ + 255) / 256), 256, 0, stream>>>(Wproj, wproj16, (long)D_ * D_);
  cvt_f16<<<(int)(((long)D_ * HID_ + 255) / 256), 256, 0, stream>>>(W1, w1_16, (long)D_ * HID_);
  cvt_f16<<<(int)(((long)HID_ * D_ + 255) / 256), 256, 0, stream>>>(W2, w2_16, (long)HID_ * D_);

  // 2. QKV GEMM: [4096,2304] = [4096,768] @ [768,2304] + bqkv
  gemm_wmma<0><<<dim3(3 * D_ / 128, BL / 64, 1), 256, 0, stream>>>(
      h16, wqkv16, bqkv, nullptr, qkvF, (int)BL, 3 * D_, D_, 0, 0, 0);

  // 3. split + mask prep
  split_qkv<<<(int)((BL * D_ + 255) / 256), 256, 0, stream>>>(qkvF, q16, k16, v16);
  prep_logts<<<(int)((BL + 255) / 256), 256, 0, stream>>>(ts, lts, (int)BL);

  // 4. windowed flash attention
  attn_kernel<<<dim3(L_ / 16, H_, B_), 32, 0, stream>>>(q16, k16, v16, lts, kpm, attn16);

  // 5. proj GEMM + residual -> xm
  gemm_wmma<1><<<dim3(D_ / 128, BL / 64, 1), 256, 0, stream>>>(
      attn16, wproj16, bproj, x, xmF, (int)BL, D_, D_, 0, 0, 0);

  // 6. ToMe: scores + selection + merge
  kmean_norm<<<(int)BL, 64, 0, stream>>>(qkvF, a16, bT16);
  gemm_wmma<0><<<dim3(TA_ / 128, TA_ / 64, B_), 256, 0, stream>>>(
      a16, bT16, nullptr, nullptr, scores, TA_, TA_, HD_,
      (long)TA_ * HD_, (long)HD_ * TA_, (long)TA_ * TA_);
  set_prot<<<(B_ * TA_ + 255) / 256, 256, 0, stream>>>(scores);
  rowargmax<<<B_ * TA_, 256, 0, stream>>>(scores, nmax, nidx);
  tome_sort<<<B_, 1024, 0, stream>>>(nmax, nidx, srcI, dstI, unmI);
  merge_init<<<dim3(L2_, B_), 256, 0, stream>>>(xmF, ts, unmI, xsum, snew);
  merge_scatter<<<dim3(R_, B_), 256, 0, stream>>>(xmF, ts, srcI, dstI, xsum, snew);
  merge_final<<<B_ * L2_, 256, 0, stream>>>(xsum, snew);  // xsum now = x_new

  // 7. LN2 + MLP
  ln_f16<<<B_ * L2_, 256, 0, stream>>>(xsum, ln2g, ln2b, h2_16, D_);
  gemm_wmma<2><<<dim3(HID_ / 128, B_ * L2_ / 64, 1), 256, 0, stream>>>(
      h2_16, w1_16, b1, nullptr, mlp16, B_ * L2_, HID_, D_, 0, 0, 0);
  gemm_wmma<1><<<dim3(D_ / 128, B_ * L2_ / 64, 1), 256, 0, stream>>>(
      mlp16, w2_16, b2, xsum, d_out, B_ * L2_, D_, HID_, 0, 0, 0);
}
